// Decoder_21852793602802
// MI455X (gfx1250) — matrix operand order repaired and checked by reading, not compiled, not run on hardware
//
#include <hip/hip_runtime.h>
#include <math.h>

// ---------------------------------------------------------------------------
// DA-RNN decoder for MI455X (gfx1250), compile-only optimized build.
//
// Structure:
//   encP[b,t,:] = IE[b,t,:] @ attn_w1[512:768,:] + attn_b1   (step-invariant,
//   hoisted; computed once with v_wmma_f32_16x16x32_bf16, stored bf16).
//
//   The 128-step scan runs in ONE kernel launch: 128 blocks x 512 threads,
//   4 batch rows per block (batch rows are fully independent -> no global
//   sync). Weights are pre-converted to bf16 and each weight load feeds 4
//   batch-row FMAs, cutting the dominant L2 weight re-read traffic 8x vs a
//   1-row/block fp32 layout. All state (h,c) lives in LDS as [k][4] float4.
//
// Workspace (<= ~68 MB, L2-resident on the 192 MB L2):
//   [0)        encP  bf16  B*T*E           = 33.5 MB
//   [33554432) IEbf  bf16  B*T*E           = 33.5 MB
//   [67108864) W1t   bf16  [256 e][512 k]  = 0.25 MB (transposed hc-part)
//   [67371008) WHHbf bf16  [1024][256]     = 0.5 MB
// ---------------------------------------------------------------------------

#define BB   512
#define TT   128
#define EE   256
#define DD   256
#define RR   4          // batch rows per block in the scan kernel
#define NT   512        // threads per scan block (16 wave32 waves)

typedef __attribute__((ext_vector_type(16))) __bf16 v16bf;
typedef __attribute__((ext_vector_type(8)))  float  v8f;
typedef unsigned short u16t;
typedef unsigned int   u32t;

__device__ __forceinline__ float fast_tanh(float x) {
#if __has_builtin(__builtin_amdgcn_tanhf)
  return __builtin_amdgcn_tanhf(x);     // gfx1250 v_tanh_f32
#else
  return tanhf(x);
#endif
}
__device__ __forceinline__ float sigf(float x) { return 1.0f / (1.0f + __expf(-x)); }

__device__ __forceinline__ u16t f2bf(float f) {        // RNE float->bf16
  u32t x = __float_as_uint(f);
  u32t r = x + 0x7fffu + ((x >> 16) & 1u);
  return (u16t)(r >> 16);
}
__device__ __forceinline__ float bf2f(u16t u) { return __uint_as_float(((u32t)u) << 16); }
__device__ __forceinline__ float bflo(u32t u) { return __uint_as_float(u << 16); }
__device__ __forceinline__ float bfhi(u32t u) { return __uint_as_float(u & 0xffff0000u); }

// ---------------------------------------------------------------------------
// Prologue converters (one-time, HBM-bandwidth trivial).
// ---------------------------------------------------------------------------
__global__ void __launch_bounds__(256)
cvt_bf16(const float* __restrict__ src, u16t* __restrict__ dst, int n) {
  int i = blockIdx.x * 256 + threadIdx.x;
  if (i < n) dst[i] = f2bf(src[i]);
}

// W1 hc-part [512 k][256 e] -> transposed bf16 [256 e][512 k]
__global__ void __launch_bounds__(256)
cvt_w1t(const float* __restrict__ w1, u16t* __restrict__ w1t) {
  int n = blockIdx.x * 256 + threadIdx.x;        // n < 131072
  int e = n >> 9, kk = n & 511;
  w1t[n] = f2bf(w1[(size_t)kk * EE + e]);
}

// ---------------------------------------------------------------------------
// Kernel 1: encP = bf16(IE) @ bf16(W1_enc) + b1, fp32 accumulate, bf16 out.
// Grid (EE/16, M/64), 128 threads (4 waves, one 16x16 tile each).
// ---------------------------------------------------------------------------
__global__ void __launch_bounds__(128)
encproj_wmma(const float* __restrict__ ie,      // [B*T, E]
             const float* __restrict__ w1,      // [768, 256]
             const float* __restrict__ b1,      // [256]
             u16t* __restrict__ encp) {         // [B*T, E] bf16
  const int lane = threadIdx.x & 31;
  const int wave = threadIdx.x >> 5;
  const int lmod = lane & 15;
  const int lhi  = lane >> 4;

  const int nBase = blockIdx.x * 16;
  const int mBase = blockIdx.y * 64 + wave * 16;
  const int m     = mBase + lmod;

  v8f acc = {};
#pragma unroll 2
  for (int kb = 0; kb < EE; kb += 32) {
    v16bf a, b;
#pragma unroll
    for (int j = 0; j < 8; ++j) {
      // A bf16 16x32: VGPR j holds K pair; lanes>=16 shift K by +8.
      const int ka = kb + (j < 4 ? 2 * j : 16 + 2 * (j - 4)) + 8 * lhi;
      const float2 av = *(const float2*)(ie + (size_t)m * EE + ka);
      a[2 * j]     = (__bf16)av.x;
      a[2 * j + 1] = (__bf16)av.y;
      // B bf16 32x16: n = lane%16, K same order as A.
      const int kbb = kb + (j < 4 ? 2 * j : 16 + 2 * (j - 4)) + 8 * lhi;
      b[2 * j]     = (__bf16)w1[(size_t)(512 + kbb)     * EE + nBase + lmod];
      b[2 * j + 1] = (__bf16)w1[(size_t)(512 + kbb + 1) * EE + nBase + lmod];
    }
    acc = __builtin_amdgcn_wmma_f32_16x16x32_bf16(
        false, a, false, b, (short)0, acc, false, false);
  }
#pragma unroll
  for (int r = 0; r < 8; ++r) {
    const int mm = mBase + r + 8 * lhi;       // C/D: m = r + 8*(lane/16)
    const int nn = nBase + lmod;
    encp[(size_t)mm * EE + nn] = f2bf(acc[r] + b1[nn]);
  }
}

// ---------------------------------------------------------------------------
// Kernel 2: full 128-step scan. 128 blocks, RR=4 batch rows each, 512 thr.
// ---------------------------------------------------------------------------
__global__ void __launch_bounds__(NT)
decoder_scan(const u16t*  __restrict__ encp,   // [B,T,E] bf16
             const u16t*  __restrict__ iebf,   // [B,T,E] bf16
             const u16t*  __restrict__ w1t,    // [256 e][512 k] bf16
             const u16t*  __restrict__ whhb,   // [1024][256] bf16
             const float* __restrict__ yh,     // [B,T]
             const float* __restrict__ w2,     // [256]
             const float* __restrict__ b2,     // [1]
             const float* __restrict__ wih,    // [1024]
             const float* __restrict__ bih,    // [1024]
             const float* __restrict__ bhh,    // [1024]
             const float* __restrict__ fcw,    // [257]
             const float* __restrict__ fcb,    // [1]
             const float* __restrict__ fcfw,   // [512]
             const float* __restrict__ fcfb,   // [1]
             float* __restrict__ out) {        // [B]
  const int tid  = threadIdx.x;
  const int b0   = blockIdx.x * RR;
  const int e    = tid & 255;                  // hidden/feature index
  const int half = tid >> 8;                   // 0/1 (wave-uniform)
  const int lane = tid & 31;
  const int wv   = tid >> 5;                   // 0..15

  __shared__ float4 sh4[DD];                   // h  as [k][4 rows]
  __shared__ float4 sc4[DD];                   // c  as [k][4 rows]
  __shared__ float  shcp[RR * EE];             // hc @ W1_hc, [r][e]
  __shared__ float  sp[2 * RR * EE];           // partials / generic scratch
  __shared__ float  sctx[RR * EE];             // context, [r][e]
  __shared__ float  sscore[RR * TT];           // attention, [r][t]
  __shared__ float  sred[NT];                  // softmax scratch
  __shared__ float  spre[4 * RR * EE];         // gate preacts [gate][r][e]
  __shared__ float  sw2[EE], sfcw[EE];
  __shared__ float  swih[4 * DD], sbias[4 * DD];
  __shared__ float  sty[RR];                   // y_tilde per row

  float* shf = (float*)sh4;
  float* scf = (float*)sc4;
  shf[tid] = 0.0f; shf[tid + NT] = 0.0f;
  scf[tid] = 0.0f; scf[tid + NT] = 0.0f;
  if (tid < EE) { sw2[tid] = w2[tid]; sfcw[tid] = fcw[tid]; }
  swih[tid]       = wih[tid];
  swih[tid + NT]  = wih[tid + NT];
  sbias[tid]      = bih[tid]      + bhh[tid];
  sbias[tid + NT] = bih[tid + NT] + bhh[tid + NT];
  __syncthreads();

  const float fcw_y = fcw[EE];
  const float fcb0  = fcb[0];
  const float b2v   = b2[0];

  for (int t = 0; t < TT; ++t) {
    // ---- hcp[r][e] = sum_k h[r][k]*W1[k,e] + c[r][k]*W1[256+k,e]
    // half 0 handles the h term, half 1 the c term; combine after.
    {
      const u16t* wrow = w1t + (size_t)e * 512 + half * 256;  // contiguous k
      __builtin_prefetch(wrow, 0, 0);
      const float4* hb = half ? sc4 : sh4;
      float a0 = 0.f, a1 = 0.f, a2 = 0.f, a3 = 0.f;
#pragma unroll 2
      for (int k = 0; k < 256; k += 8) {
        const uint4 wq = *(const uint4*)(wrow + k);
        const u32t wu[4] = {wq.x, wq.y, wq.z, wq.w};
#pragma unroll
        for (int j = 0; j < 4; ++j) {
          const float w0 = bflo(wu[j]), w1v = bfhi(wu[j]);
          const float4 h0 = hb[k + 2 * j];
          const float4 h1 = hb[k + 2 * j + 1];
          a0 += h0.x * w0 + h1.x * w1v;
          a1 += h0.y * w0 + h1.y * w1v;
          a2 += h0.z * w0 + h1.z * w1v;
          a3 += h0.w * w0 + h1.w * w1v;
        }
      }
      sp[half * 1024 + 0 * EE + e] = a0;
      sp[half * 1024 + 1 * EE + e] = a1;
      sp[half * 1024 + 2 * EE + e] = a2;
      sp[half * 1024 + 3 * EE + e] = a3;
    }
    __syncthreads();
    shcp[tid]       = sp[tid]       + sp[1024 + tid];
    shcp[tid + NT]  = sp[tid + NT]  + sp[1024 + tid + NT];
    __syncthreads();

    // ---- scores[r][t'] = tanh(encP + hcp) . w2  (wave32 shuffle reduce)
    for (int i = 0; i < 32; ++i) {
      const int p  = wv * 32 + i;              // 512 (row,t') pairs
      const int rl = p >> 7, tt = p & 127;
      const u16t* ep = encp + ((size_t)(b0 + rl) * TT + tt) * EE;
      float s = 0.0f;
#pragma unroll
      for (int j = 0; j < 8; ++j) {
        const int ee = lane + 32 * j;          // coalesced 32-wide
        s += fast_tanh(bf2f(ep[ee]) + shcp[rl * EE + ee]) * sw2[ee];
      }
#pragma unroll
      for (int off = 16; off; off >>= 1) s += __shfl_xor(s, off, 32);
      if (lane == 0) sscore[p] = s + b2v;
    }
    __syncthreads();

    // ---- row-wise softmax over T=128 (4 rows x 128 = 512 = NT)
    const int ti = tid & 127;
    const int rowb = tid & ~127;
    float v = sscore[tid];
    sred[tid] = v;
    __syncthreads();
    for (int off = 64; off >= 1; off >>= 1) {
      if (ti < off) { float o = sred[tid + off]; if (o > sred[tid]) sred[tid] = o; }
      __syncthreads();
    }
    const float maxv = sred[rowb];
    __syncthreads();
    const float ev = __expf(v - maxv);
    sred[tid] = ev;
    __syncthreads();
    for (int off = 64; off >= 1; off >>= 1) {
      if (ti < off) sred[tid] += sred[tid + off];
      __syncthreads();
    }
    const float inv = 1.0f / sred[rowb];
    __syncthreads();
    sscore[tid] = ev * inv;
    __syncthreads();

    // ---- ctx[r][e] = sum_t attn[r][t] * IE[b0+r,t,e]; y_tilde partials
    for (int j = 0; j < 2; ++j) {
      const int rl = half * 2 + j;
      const u16t* ib = iebf + ((size_t)(b0 + rl) * TT) * EE + e;
      float cx = 0.0f;
#pragma unroll 4
      for (int tt = 0; tt < TT; ++tt) cx += sscore[rl * TT + tt] * bf2f(ib[(size_t)tt * EE]);
      sctx[rl * EE + e] = cx;
      sp[rl * EE + e]   = cx * sfcw[e];
    }
    __syncthreads();
    {   // segmented reduce: 4 segments of 256 -> sp[r*256]
      const int seg = tid >> 7;
      sp[seg * EE + ti] += sp[seg * EE + ti + 128];
      __syncthreads();
      for (int off = 64; off >= 1; off >>= 1) {
        if (ti < off) sp[seg * EE + ti] += sp[seg * EE + ti + off];
        __syncthreads();
      }
    }
    if (tid < RR) sty[tid] = sp[tid * EE] + yh[(size_t)(b0 + tid) * TT + t] * fcw_y + fcb0;
    __syncthreads();

    // ---- LSTM gates: thread (e,half) owns gate rows {2h, 2h+1} for unit e,
    //      each weight load feeds 4 batch-row FMAs.
    {
      const int ga = 2 * half;                 // gate types: 0,1 or 2,3
      const u16t* wa = whhb + (size_t)(ga * DD + e) * DD;        // contiguous k
      const u16t* wb = whhb + (size_t)((ga + 1) * DD + e) * DD;
      __builtin_prefetch(wa, 0, 0);
      __builtin_prefetch(wb, 0, 0);
      float aA0 = 0.f, aA1 = 0.f, aA2 = 0.f, aA3 = 0.f;
      float aB0 = 0.f, aB1 = 0.f, aB2 = 0.f, aB3 = 0.f;
#pragma unroll 2
      for (int k = 0; k < 256; k += 8) {
        const uint4 qa = *(const uint4*)(wa + k);
        const uint4 qb = *(const uint4*)(wb + k);
        const u32t ua[4] = {qa.x, qa.y, qa.z, qa.w};
        const u32t ub[4] = {qb.x, qb.y, qb.z, qb.w};
#pragma unroll
        for (int j = 0; j < 4; ++j) {
          const float4 h0 = sh4[k + 2 * j];
          const float4 h1 = sh4[k + 2 * j + 1];
          const float wa0 = bflo(ua[j]), wa1 = bfhi(ua[j]);
          const float wb0 = bflo(ub[j]), wb1 = bfhi(ub[j]);
          aA0 += h0.x * wa0 + h1.x * wa1;  aB0 += h0.x * wb0 + h1.x * wb1;
          aA1 += h0.y * wa0 + h1.y * wa1;  aB1 += h0.y * wb0 + h1.y * wb1;
          aA2 += h0.z * wa0 + h1.z * wa1;  aB2 += h0.z * wb0 + h1.z * wb1;
          aA3 += h0.w * wa0 + h1.w * wa1;  aB3 += h0.w * wb0 + h1.w * wb1;
        }
      }
      const float wiA = swih[ga * DD + e],  bA = sbias[ga * DD + e];
      const float wiB = swih[(ga + 1) * DD + e], bB = sbias[(ga + 1) * DD + e];
      spre[ga * 1024 + 0 * EE + e] = sty[0] * wiA + bA + aA0;
      spre[ga * 1024 + 1 * EE + e] = sty[1] * wiA + bA + aA1;
      spre[ga * 1024 + 2 * EE + e] = sty[2] * wiA + bA + aA2;
      spre[ga * 1024 + 3 * EE + e] = sty[3] * wiA + bA + aA3;
      spre[(ga + 1) * 1024 + 0 * EE + e] = sty[0] * wiB + bB + aB0;
      spre[(ga + 1) * 1024 + 1 * EE + e] = sty[1] * wiB + bB + aB1;
      spre[(ga + 1) * 1024 + 2 * EE + e] = sty[2] * wiB + bB + aB2;
      spre[(ga + 1) * 1024 + 3 * EE + e] = sty[3] * wiB + bB + aB3;
    }
    __syncthreads();                           // all h reads + spre writes done

    // ---- pointwise LSTM update; thread (e,half) owns rows {2h, 2h+1}
#pragma unroll
    for (int j = 0; j < 2; ++j) {
      const int rl = half * 2 + j;
      const float pi = spre[0 * 1024 + rl * EE + e];
      const float pf = spre[1 * 1024 + rl * EE + e];
      const float pg = spre[2 * 1024 + rl * EE + e];
      const float po = spre[3 * 1024 + rl * EE + e];
      const float cold = scf[e * 4 + rl];
      const float cn = sigf(pf) * cold + sigf(pi) * fast_tanh(pg);
      const float hn = sigf(po) * fast_tanh(cn);
      scf[e * 4 + rl] = cn;                    // (r,e) owned exclusively
      shf[e * 4 + rl] = hn;
    }
    __syncthreads();
  }

  // ---- out[b0+r] = concat(h, ctx) @ fcf_w + fcf_b
  for (int j = 0; j < 2; ++j) {
    const int rl = half * 2 + j;
    sp[rl * EE + e] = shf[e * 4 + rl] * fcfw[e] + sctx[rl * EE + e] * fcfw[EE + e];
  }
  __syncthreads();
  {
    const int ti = tid & 127;
    const int seg = tid >> 7;
    sp[seg * EE + ti] += sp[seg * EE + ti + 128];
    __syncthreads();
    for (int off = 64; off >= 1; off >>= 1) {
      if (ti < off) sp[seg * EE + ti] += sp[seg * EE + ti + off];
      __syncthreads();
    }
  }
  if (tid < RR) out[b0 + tid] = sp[tid * EE] + fcfb[0];
}

// ---------------------------------------------------------------------------
extern "C" void kernel_launch(void* const* d_in, const int* in_sizes, int n_in,
                              void* d_out, int out_size, void* d_ws, size_t ws_size,
                              hipStream_t stream) {
  const float* ie   = (const float*)d_in[0];
  const float* yh   = (const float*)d_in[1];
  const float* aw1  = (const float*)d_in[2];
  const float* ab1  = (const float*)d_in[3];
  const float* aw2  = (const float*)d_in[4];
  const float* ab2  = (const float*)d_in[5];
  const float* wih  = (const float*)d_in[6];
  const float* whh  = (const float*)d_in[7];
  const float* bih  = (const float*)d_in[8];
  const float* bhh  = (const float*)d_in[9];
  const float* fcw  = (const float*)d_in[10];
  const float* fcb  = (const float*)d_in[11];
  const float* fcfw = (const float*)d_in[12];
  const float* fcfb = (const float*)d_in[13];

  char* ws = (char*)d_ws;
  u16t* encp  = (u16t*)(ws);                         // 33,554,432 B
  u16t* iebf  = (u16t*)(ws + 33554432);              // 33,554,432 B
  u16t* w1t   = (u16t*)(ws + 67108864);              //    262,144 B
  u16t* whhb  = (u16t*)(ws + 67371008);              //    524,288 B

  const int nIE = BB * TT * EE;                      // 16,777,216
  cvt_bf16<<<nIE / 256, 256, 0, stream>>>(ie, iebf, nIE);
  cvt_w1t <<<(512 * EE) / 256, 256, 0, stream>>>(aw1, w1t);
  cvt_bf16<<<(4 * DD * DD) / 256, 256, 0, stream>>>(whh, whhb, 4 * DD * DD);

  dim3 g1(EE / 16, (BB * TT) / 64);
  encproj_wmma<<<g1, 128, 0, stream>>>(ie, aw1, ab1, encp);

  decoder_scan<<<BB / RR, NT, 0, stream>>>(encp, iebf, w1t, whhb,
                                           yh, aw2, ab2, wih, bih, bhh,
                                           fcw, fcb, fcfw, fcfb,
                                           (float*)d_out);
}
